// DualStreamBlock_78623671320688
// MI455X (gfx1250) — compile-verified, matrix operand order checked by reading, not run on hardware
//
#include <hip/hip_runtime.h>
#include <hip/hip_bf16.h>
#include <math.h>

// ---------------- types for WMMA ----------------
typedef __bf16 bf16;
typedef __attribute__((ext_vector_type(16))) __bf16 v16bf;
typedef __attribute__((ext_vector_type(8)))  __bf16 v8bf;
typedef __attribute__((ext_vector_type(8)))  float  v8f;

union ABFrag { v16bf v; v8bf h[2]; };

#define C_DIM   1024
#define HEADS   16
#define HD      64
#define RANK    16
#define INTER_D 2816
#define BB      2
#define NN      2048
#define ROWS    (BB*NN)      // 4096
#define SIXC    (6*C_DIM)
#define LN_EPS  1e-5f

__device__ __forceinline__ float sigmoidf_(float x){ return 1.0f/(1.0f+__expf(-x)); }
__device__ __forceinline__ float siluf_(float x){ return x*sigmoidf_(x); }

// =====================================================================
// Generic bf16 WMMA GEMM:  C[M,N] = A[M,K] * B[K,N],  B given transposed
// (Bt is [N,K] row-major).  Block tile 128M x 64N, 8 waves (16M each).
// B tile (64 x 32K, 4KB) is staged into LDS with global_load_async_to_lds
// (ASYNCcnt path), double-buffered, shared by all 8 waves.
// =====================================================================
__global__ __launch_bounds__(256)
void gemm_bf16_wmma(const bf16* __restrict__ A, const bf16* __restrict__ Bt,
                    float* __restrict__ Cf, bf16* __restrict__ Cb,
                    int M, int N, int K, int lda, int ldb, int ldc)
{
    __shared__ __align__(16) bf16 btile[2][64 * 32];   // 2 x 4KB double buffer

    const int tid  = threadIdx.x;
    const int lane = tid & 31;
    const int wave = tid >> 5;      // 0..7
    const int half = lane >> 4;
    const int lr   = lane & 15;

    const int m0 = blockIdx.y * 128 + wave * 16;
    const int n0 = blockIdx.x * 64;

    // B staging map: thread -> (col 0..63, 16B chunk 0..3)
    const int scol = tid >> 2;
    const int schk = tid & 3;
    const int gcol = min(n0 + scol, N - 1);
    const bf16* sb_g = Bt + (size_t)gcol * ldb + schk * 8;
    const unsigned sb_l0 = (unsigned)(uintptr_t)(&btile[0][scol * 32 + schk * 8]);
    const unsigned sb_l1 = (unsigned)(uintptr_t)(&btile[1][scol * 32 + schk * 8]);

    const int arow = min(m0 + lr, M - 1);
    const bf16* ap = A + (size_t)arow * lda + half * 8;

    v8f acc[4] = {v8f{0}, v8f{0}, v8f{0}, v8f{0}};

    // prologue: async-fill buffer 0 with K-step 0
    asm volatile("global_load_async_to_lds_b128 %0, %1, off"
                 :: "v"(sb_l0), "v"(sb_g) : "memory");

    const int steps = K >> 5;
    for (int i = 0; i < steps; ++i) {
        const int k = i << 5;
        // current buffer's async fill (issued last iter) complete for this wave
        asm volatile("s_wait_asynccnt 0x0" ::: "memory");
        // all waves' fills visible + previous compute finished everywhere
        __syncthreads();
        if (i + 1 < steps) {
            unsigned dst = ((i + 1) & 1) ? sb_l1 : sb_l0;
            asm volatile("global_load_async_to_lds_b128 %0, %1, off"
                         :: "v"(dst), "v"(sb_g + k + 32) : "memory");
        }
        const bf16* bt = btile[i & 1];

        __builtin_prefetch(ap + k + 128, 0, 1);
        ABFrag a;
        a.h[0] = *(const v8bf*)(ap + k);
        a.h[1] = *(const v8bf*)(ap + k + 16);

        // load all four B fragments up front (distinct regs) so the LDS
        // loads issue as a clause and WMMAs aren't serialized on dscnt==0
        ABFrag bf4[4];
#pragma unroll
        for (int j = 0; j < 4; ++j) {
            bf4[j].h[0] = *(const v8bf*)(&bt[(j * 16 + lr) * 32 + half * 8]);
            bf4[j].h[1] = *(const v8bf*)(&bt[(j * 16 + lr) * 32 + 16 + half * 8]);
        }
#pragma unroll
        for (int j = 0; j < 4; ++j) {
            acc[j] = __builtin_amdgcn_wmma_f32_16x16x32_bf16(
                false, a.v, false, bf4[j].v, (short)0, acc[j], false, false);
        }
    }

#pragma unroll
    for (int j = 0; j < 4; ++j) {
        int col = n0 + j * 16 + lr;
        if (col >= N) continue;
#pragma unroll
        for (int r = 0; r < 8; ++r) {
            int row = m0 + half * 8 + r;
            if (row < M) {
                if (Cb) Cb[(size_t)row * ldc + col] = (bf16)acc[j][r];
                else    Cf[(size_t)row * ldc + col] = acc[j][r];
            }
        }
    }
}

// =====================================================================
// Flash attention, one wave = one 16-query tile of one (b,h).
// Q,K: [B*H, N, 64] bf16   Vt: [B*H, 64, N] bf16   Out: [B, N, C] bf16
// =====================================================================
__global__ __launch_bounds__(128)
void flash_attn_wmma(const bf16* __restrict__ Qh, const bf16* __restrict__ Kh,
                     const bf16* __restrict__ Vt, bf16* __restrict__ Out)
{
    __shared__ __align__(16) bf16 lds[4][16 * 32];

    const int lane = threadIdx.x & 31;
    const int wave = threadIdx.x >> 5;
    const int half = lane >> 4;
    const int lr   = lane & 15;

    const int g  = blockIdx.x * 4 + wave;   // global wave id over B*H*(N/16)
    const int qt = g & 127;                 // query tile (N/16 = 128)
    const int bh = g >> 7;                  // 0..31
    const int b  = bh >> 4, h = bh & 15;

    const bf16* qb  = Qh + (size_t)bh * NN * HD;
    const bf16* kbp = Kh + (size_t)bh * NN * HD;
    const bf16* vb  = Vt + (size_t)bh * HD * NN;
    bf16* ob = Out + (size_t)b * NN * C_DIM;

    // Q fragments (16 rows x 64 deep -> two K=32 fragments)
    ABFrag aq0, aq1;
    {
        const bf16* qp = qb + (size_t)(qt * 16 + lr) * HD + half * 8;
        aq0.h[0] = *(const v8bf*)(qp);      aq0.h[1] = *(const v8bf*)(qp + 16);
        aq1.h[0] = *(const v8bf*)(qp + 32); aq1.h[1] = *(const v8bf*)(qp + 48);
    }

    float mrow[8], lrow[8], alpha[8];
#pragma unroll
    for (int r = 0; r < 8; ++r) { mrow[r] = -1e30f; lrow[r] = 0.0f; }
    v8f o[4] = {v8f{0}, v8f{0}, v8f{0}, v8f{0}};
    bf16* myLds = lds[wave];

    for (int kb = 0; kb < NN; kb += 32) {
        v8f s[2] = {v8f{0}, v8f{0}};
#pragma unroll
        for (int t = 0; t < 2; ++t) {
            const bf16* kp = kbp + (size_t)(kb + t * 16 + lr) * HD + half * 8;
            ABFrag bk;
            bk.h[0] = *(const v8bf*)(kp);      bk.h[1] = *(const v8bf*)(kp + 16);
            s[t] = __builtin_amdgcn_wmma_f32_16x16x32_bf16(
                false, aq0.v, false, bk.v, (short)0, s[t], false, false);
            bk.h[0] = *(const v8bf*)(kp + 32); bk.h[1] = *(const v8bf*)(kp + 48);
            s[t] = __builtin_amdgcn_wmma_f32_16x16x32_bf16(
                false, aq1.v, false, bk.v, (short)0, s[t], false, false);
        }
        // online softmax, rows live in 16-lane half-groups
#pragma unroll
        for (int r = 0; r < 8; ++r) {
            float s0 = s[0][r] * 0.125f;   // D^-0.5 = 1/8
            float s1 = s[1][r] * 0.125f;
            float mx = fmaxf(s0, s1);
#pragma unroll
            for (int d = 8; d >= 1; d >>= 1) mx = fmaxf(mx, __shfl_xor(mx, d, 32));
            float nm = fmaxf(mrow[r], mx);
            float al = __expf(mrow[r] - nm);
            mrow[r] = nm;
            float p0 = __expf(s0 - nm), p1 = __expf(s1 - nm);
            float ps = p0 + p1;
#pragma unroll
            for (int d = 8; d >= 1; d >>= 1) ps += __shfl_xor(ps, d, 32);
            lrow[r] = lrow[r] * al + ps;
            alpha[r] = al;
            int row = half * 8 + r;
            myLds[row * 32 + lr]      = (bf16)p0;
            myLds[row * 32 + 16 + lr] = (bf16)p1;
        }
#pragma unroll
        for (int j = 0; j < 4; ++j)
#pragma unroll
            for (int r = 0; r < 8; ++r) o[j][r] *= alpha[r];

        asm volatile("s_wait_dscnt 0" ::: "memory");

        ABFrag pa;   // re-read P in A-fragment layout
        pa.h[0] = *(const v8bf*)(&myLds[lr * 32 + half * 8]);
        pa.h[1] = *(const v8bf*)(&myLds[lr * 32 + 16 + half * 8]);
#pragma unroll
        for (int j = 0; j < 4; ++j) {
            const bf16* vp = vb + (size_t)(j * 16 + lr) * NN + kb + half * 8;
            ABFrag bv;
            bv.h[0] = *(const v8bf*)(vp);
            bv.h[1] = *(const v8bf*)(vp + 16);
            o[j] = __builtin_amdgcn_wmma_f32_16x16x32_bf16(
                false, pa.v, false, bv.v, (short)0, o[j], false, false);
        }
        asm volatile("s_wait_dscnt 0" ::: "memory");
    }

#pragma unroll
    for (int j = 0; j < 4; ++j) {
        int col = h * HD + j * 16 + lr;
#pragma unroll
        for (int r = 0; r < 8; ++r) {
            int n = qt * 16 + half * 8 + r;
            ob[(size_t)n * C_DIM + col] = (bf16)(o[j][r] / lrow[r]);
        }
    }
}

// =====================================================================
// adaLN mods = silu(c) @ ada_w + ada_b     (2 x 6144, tiny)
// =====================================================================
__global__ void mods_kernel(const float* __restrict__ c, const float* __restrict__ W,
                            const float* __restrict__ bias, float* __restrict__ mods)
{
    int idx = blockIdx.x * 256 + threadIdx.x;
    if (idx >= BB * SIXC) return;
    int b = idx / SIXC, col = idx - b * SIXC;
    const float* cr = c + (size_t)b * C_DIM;
    float acc = bias[col];
    for (int k = 0; k < C_DIM; ++k)
        acc += siluf_(cr[k]) * W[(size_t)k * SIXC + col];
    mods[idx] = acc;
}

// =====================================================================
// Weight convert + transpose: W[K,N] f32 -> Wt[N,Kp] bf16 (zero pad K->Kp)
// =====================================================================
__global__ void convT_kernel(const float* __restrict__ W, bf16* __restrict__ Wt,
                             int K, int N, int Kp)
{
    size_t idx = (size_t)blockIdx.x * 256 + threadIdx.x;
    if (idx >= (size_t)N * Kp) return;
    int n = (int)(idx / Kp), k = (int)(idx - (size_t)n * Kp);
    Wt[idx] = (bf16)((k < K) ? W[(size_t)k * N + n] : 0.0f);
}

__global__ void fillz_bf16_kernel(bf16* __restrict__ p, size_t n)
{
    size_t idx = (size_t)blockIdx.x * 256 + threadIdx.x;
    if (idx < n) p[idx] = (bf16)0.0f;
}

// =====================================================================
// LayerNorm (+ optional modulate) -> bf16 outputs
// =====================================================================
__global__ __launch_bounds__(256)
void ln_mod_kernel(const float* __restrict__ X, const float* __restrict__ gw,
                   const float* __restrict__ gb, const float* __restrict__ mods,
                   int shOfs, int scOfs,
                   bf16* __restrict__ xnOut, bf16* __restrict__ modOut)
{
    __shared__ float red[256];
    const int row = blockIdx.x;
    const int b = row >> 11;               // 2048 rows per batch
    const float* xr = X + (size_t)row * C_DIM;

    float s = 0.0f;
    for (int c = threadIdx.x; c < C_DIM; c += 256) s += xr[c];
    red[threadIdx.x] = s; __syncthreads();
    for (int st = 128; st > 0; st >>= 1) {
        if (threadIdx.x < st) red[threadIdx.x] += red[threadIdx.x + st];
        __syncthreads();
    }
    float mean = red[0] * (1.0f / C_DIM);
    __syncthreads();

    float v = 0.0f;
    for (int c = threadIdx.x; c < C_DIM; c += 256) { float d = xr[c] - mean; v += d * d; }
    red[threadIdx.x] = v; __syncthreads();
    for (int st = 128; st > 0; st >>= 1) {
        if (threadIdx.x < st) red[threadIdx.x] += red[threadIdx.x + st];
        __syncthreads();
    }
    float inv = rsqrtf(red[0] * (1.0f / C_DIM) + LN_EPS);

    const float* sh = mods + (size_t)b * SIXC + shOfs;
    const float* sc = mods + (size_t)b * SIXC + scOfs;
    for (int c = threadIdx.x; c < C_DIM; c += 256) {
        float xn = (xr[c] - mean) * inv * gw[c] + gb[c];
        if (xnOut)  xnOut[(size_t)row * C_DIM + c] = (bf16)xn;
        if (modOut) modOut[(size_t)row * C_DIM + c] = (bf16)(xn * (1.0f + sc[c]) + sh[c]);
    }
}

// =====================================================================
// DoRA prep: q += mag[h]*delta ; split qkv -> Q,K (head-major), V^T
// =====================================================================
__global__ void dora_prep_kernel(const float* __restrict__ qkv, const float* __restrict__ delta,
                                 const float* __restrict__ mag,
                                 bf16* __restrict__ Qh, bf16* __restrict__ Kh,
                                 bf16* __restrict__ Vt)
{
    size_t idx = (size_t)blockIdx.x * 256 + threadIdx.x;
    if (idx >= (size_t)ROWS * C_DIM) return;
    int bn = (int)(idx >> 10);
    int cc = (int)(idx & 1023);
    int h = cc >> 6, d = cc & 63;
    int b = bn >> 11, n = bn & 2047;

    const float* qr = qkv + (size_t)bn * (3 * C_DIM);
    float q = qr[cc] + mag[h] * delta[(size_t)bn * C_DIM + cc];
    float k = qr[C_DIM + cc];
    float v = qr[2 * C_DIM + cc];

    size_t bh = (size_t)b * HEADS + h;
    Qh[bh * NN * HD + (size_t)n * HD + d] = (bf16)q;
    Kh[bh * NN * HD + (size_t)n * HD + d] = (bf16)k;
    Vt[bh * HD * NN + (size_t)d * NN + n] = (bf16)v;
}

// =====================================================================
// HGRN gates: vf -> f = sigmoid(vf); vi -> i = silu(vi)*(1-f)   (in place)
// =====================================================================
__global__ void hgrn_fi_kernel(float* __restrict__ vf, float* __restrict__ vi)
{
    size_t idx = (size_t)blockIdx.x * 256 + threadIdx.x;
    if (idx >= (size_t)ROWS * C_DIM) return;
    float f = sigmoidf_(vf[idx]);
    float i = siluf_(vi[idx]) * (1.0f - f);
    vf[idx] = f;
    vi[idx] = i;
}

// h[n] = f[n]*h[n-1] + i[n], per (b,c) channel
__global__ void hgrn_scan_kernel(const float* __restrict__ F, const float* __restrict__ I,
                                 float* __restrict__ H)
{
    int t = blockIdx.x * 256 + threadIdx.x;
    if (t >= BB * C_DIM) return;
    int b = t >> 10, c = t & 1023;
    size_t base = (size_t)b * NN * C_DIM + c;
    float h = 0.0f;
    for (int n = 0; n < NN; ++n) {
        size_t ix = base + (size_t)n * C_DIM;
        h = F[ix] * h + I[ix];
        H[ix] = h;
    }
}

// per-head RMSNorm * gn * silu(vg) -> bf16  (one wave per (row,head))
__global__ __launch_bounds__(256)
void hgrn_out_kernel(const float* __restrict__ H, const float* __restrict__ VG,
                     const float* __restrict__ gn, bf16* __restrict__ out)
{
    int gw = blockIdx.x * 8 + (threadIdx.x >> 5);  // over ROWS*HEADS
    int lane = threadIdx.x & 31;
    int row = gw >> 4, h = gw & 15;
    size_t base = (size_t)row * C_DIM + h * HD;

    float e0 = H[base + lane];
    float e1 = H[base + 32 + lane];
    float ss = e0 * e0 + e1 * e1;
#pragma unroll
    for (int d = 16; d >= 1; d >>= 1) ss += __shfl_xor(ss, d, 32);
    float inv = rsqrtf(ss * (1.0f / HD) + LN_EPS);

    int c0 = h * HD + lane, c1 = c0 + 32;
    out[base + lane]      = (bf16)(e0 * inv * gn[c0] * siluf_(VG[base + lane]));
    out[base + 32 + lane] = (bf16)(e1 * inv * gn[c1] * siluf_(VG[base + 32 + lane]));
}

// =====================================================================
// X += g[b,:] * R   (gated residual into d_out)
// =====================================================================
__global__ void residual_kernel(float* __restrict__ X, const float* __restrict__ R,
                                const float* __restrict__ mods, int gofs)
{
    size_t idx = (size_t)blockIdx.x * 256 + threadIdx.x;
    if (idx >= (size_t)ROWS * C_DIM) return;
    int b = (int)(idx >> 21);   // NN*C = 2^21
    int c = (int)(idx & 1023);
    X[idx] += mods[(size_t)b * SIXC + gofs + c] * R[idx];
}

// silu(gate)*val  from u[ROWS, 2*INTER] -> bf16[ROWS, INTER]
__global__ void mlp_fuse_kernel(const float* __restrict__ U, bf16* __restrict__ out)
{
    size_t idx = (size_t)blockIdx.x * 256 + threadIdx.x;
    if (idx >= (size_t)ROWS * INTER_D) return;
    int row = (int)(idx / INTER_D);
    int j   = (int)(idx - (size_t)row * INTER_D);
    float g = U[(size_t)row * (2 * INTER_D) + j];
    float v = U[(size_t)row * (2 * INTER_D) + INTER_D + j];
    out[idx] = (bf16)(siluf_(g) * v);
}

// =====================================================================
// Host side
// =====================================================================
static inline size_t aln256(size_t x) { return (x + 255) & ~(size_t)255; }

static void launch_gemm(const bf16* A, const bf16* Bt, float* Cf, bf16* Cb,
                        int M, int N, int K, int lda, int ldb, int ldc,
                        hipStream_t s)
{
    dim3 grid((N + 63) / 64, (M + 127) / 128);
    gemm_bf16_wmma<<<grid, dim3(256), 0, s>>>(A, Bt, Cf, Cb, M, N, K, lda, ldb, ldc);
}

extern "C" void kernel_launch(void* const* d_in, const int* in_sizes, int n_in,
                              void* d_out, int out_size, void* d_ws, size_t ws_size,
                              hipStream_t stream)
{
    (void)in_sizes; (void)n_in; (void)out_size;

    // ---------------- workspace arena ----------------
    char* w = (char*)d_ws;
    size_t off = 0;
    auto alloc = [&](size_t bytes) -> char* {
        char* p = w + off; off = aln256(off + bytes); return p;
    };
    bf16* qkvT  = (bf16*)alloc((size_t)3 * C_DIM * C_DIM * 2);       // 3072 x 1024
    bf16* laT   = (bf16*)alloc((size_t)RANK * C_DIM * 2);            // 16 x 1024
    bf16* lbT   = (bf16*)alloc((size_t)C_DIM * 32 * 2);              // 1024 x 32 (K padded)
    bf16* projT = (bf16*)alloc((size_t)C_DIM * C_DIM * 2);
    bf16* wiT   = (bf16*)alloc((size_t)C_DIM * C_DIM * 2);
    bf16* wfT   = (bf16*)alloc((size_t)C_DIM * C_DIM * 2);
    bf16* wgT   = (bf16*)alloc((size_t)C_DIM * C_DIM * 2);
    bf16* woT   = (bf16*)alloc((size_t)C_DIM * C_DIM * 2);
    bf16* gateT = (bf16*)alloc((size_t)2 * INTER_D * C_DIM * 2);     // 5632 x 1024
    bf16* downT = (bf16*)alloc((size_t)C_DIM * INTER_D * 2);         // 1024 x 2816
    float* mods = (float*)alloc((size_t)BB * SIXC * 4);
    bf16* xn_bf  = (bf16*)alloc((size_t)ROWS * C_DIM * 2);
    bf16* mod_bf = (bf16*)alloc((size_t)ROWS * C_DIM * 2);           // msa-mod, later mlp-mod
    bf16* t1_bf  = (bf16*)alloc((size_t)ROWS * 32 * 2);
    float* deltaF = (float*)alloc((size_t)ROWS * C_DIM * 4);         // delta / Hbuf / down-out
    bf16* Qh = (bf16*)alloc((size_t)BB * HEADS * NN * HD * 2);       // also: fused mlp bf16
    bf16* Kh = (bf16*)alloc((size_t)BB * HEADS * NN * HD * 2);
    bf16* Vt = (bf16*)alloc((size_t)BB * HEADS * HD * NN * 2);
    bf16* attn_bf = (bf16*)alloc((size_t)ROWS * C_DIM * 2);          // attn-out / hgrn-out
    float* projOut = (float*)alloc((size_t)ROWS * C_DIM * 4);        // proj-out / wo-out
    float* big = (float*)alloc((size_t)ROWS * 2 * INTER_D * 4);      // qkv f32 / vi,vf,vg / mlp-u
    if (off > ws_size) return;  // not enough scratch; bail (no-op)

    // aliases into the big region (phase-reused)
    float* qkvF = big;
    float* viF  = big;
    float* vfF  = big + (size_t)ROWS * C_DIM;
    float* vgF  = big + (size_t)2 * ROWS * C_DIM;
    float* uF   = big;
    bf16* fused_bf = Qh;   // 23MB fits in Qh+Kh+Vt (24MB), consumed before reuse

    const float* cIn  = (const float*)d_in[2];
    const float* adaW = (const float*)d_in[7];
    const float* adaB = (const float*)d_in[8];

    // seed outputs with residual streams
    hipMemcpyAsync(d_out, d_in[0], (size_t)ROWS * C_DIM * 4, hipMemcpyDeviceToDevice, stream);
    hipMemcpyAsync((char*)d_out + (size_t)ROWS * C_DIM * 4, d_in[1],
                   (size_t)ROWS * C_DIM * 4, hipMemcpyDeviceToDevice, stream);

    // adaLN mods (shared by both streams)
    mods_kernel<<<(BB * SIXC + 255) / 256, 256, 0, stream>>>(cIn, adaW, adaB, mods);

    auto conv = [&](const float* W, bf16* Wt, int K, int N, int Kp) {
        size_t n = (size_t)N * Kp;
        convT_kernel<<<(unsigned)((n + 255) / 256), 256, 0, stream>>>(W, Wt, K, N, Kp);
    };
    auto elems = (size_t)ROWS * C_DIM;
    unsigned gridE = (unsigned)((elems + 255) / 256);

    for (int s = 0; s < 2; ++s) {
        const float* xin = (const float*)d_in[s == 0 ? 0 : 1];
        const float* gw  = (const float*)d_in[s == 0 ? 3 : 5];
        const float* gb  = (const float*)d_in[s == 0 ? 4 : 6];
        int dq = s == 0 ? 9 : 14;
        const float* Wqkv = (const float*)d_in[dq + 0];
        const float* mag  = (const float*)d_in[dq + 1];
        const float* Wla  = (const float*)d_in[dq + 2];
        const float* Wlb  = (const float*)d_in[dq + 3];
        const float* Wproj= (const float*)d_in[dq + 4];
        int dh = s == 0 ? 19 : 24;
        const float* Wwi = (const float*)d_in[dh + 0];
        const float* Wwf = (const float*)d_in[dh + 1];
        const float* Wwg = (const float*)d_in[dh + 2];
        const float* Wwo = (const float*)d_in[dh + 3];
        const float* gn  = (const float*)d_in[dh + 4];
        int dm = s == 0 ? 29 : 31;
        const float* Wgate = (const float*)d_in[dm + 0];
        const float* Wdown = (const float*)d_in[dm + 1];
        float* Xout = (float*)d_out + (size_t)s * ROWS * C_DIM;

        // ---- weights -> bf16 transposed ----
        conv(Wqkv,  qkvT,  C_DIM, 3 * C_DIM, C_DIM);
        conv(Wla,   laT,   C_DIM, RANK,      C_DIM);
        conv(Wlb,   lbT,   RANK,  C_DIM,     32);        // K pad 16 -> 32
        conv(Wproj, projT, C_DIM, C_DIM,     C_DIM);
        conv(Wwi,   wiT,   C_DIM, C_DIM,     C_DIM);
        conv(Wwf,   wfT,   C_DIM, C_DIM,     C_DIM);
        conv(Wwg,   wgT,   C_DIM, C_DIM,     C_DIM);
        conv(Wwo,   woT,   C_DIM, C_DIM,     C_DIM);
        conv(Wgate, gateT, C_DIM, 2 * INTER_D, C_DIM);
        conv(Wdown, downT, INTER_D, C_DIM,   INTER_D);

        // ---- LN1: xn (bf16) + msa-modulated xn (bf16) ----
        ln_mod_kernel<<<ROWS, 256, 0, stream>>>(xin, gw, gb, mods, 0, C_DIM, xn_bf, mod_bf);

        // ---- DoRA attention ----
        launch_gemm(xn_bf, qkvT, qkvF, nullptr, ROWS, 3 * C_DIM, C_DIM,
                    C_DIM, C_DIM, 3 * C_DIM, stream);
        fillz_bf16_kernel<<<(ROWS * 32 + 255) / 256, 256, 0, stream>>>(t1_bf, (size_t)ROWS * 32);
        launch_gemm(xn_bf, laT, nullptr, t1_bf, ROWS, RANK, C_DIM, C_DIM, C_DIM, 32, stream);
        launch_gemm(t1_bf, lbT, deltaF, nullptr, ROWS, C_DIM, 32, 32, 32, C_DIM, stream);
        dora_prep_kernel<<<gridE, 256, 0, stream>>>(qkvF, deltaF, mag, Qh, Kh, Vt);
        flash_attn_wmma<<<(BB * HEADS * (NN / 16)) / 4, 128, 0, stream>>>(Qh, Kh, Vt, attn_bf);
        launch_gemm(attn_bf, projT, projOut, nullptr, ROWS, C_DIM, C_DIM,
                    C_DIM, C_DIM, C_DIM, stream);
        residual_kernel<<<gridE, 256, 0, stream>>>(Xout, projOut, mods, 2 * C_DIM);

        // ---- HGRN recurrent attention (input: msa-modulated original xn) ----
        launch_gemm(mod_bf, wiT, viF, nullptr, ROWS, C_DIM, C_DIM, C_DIM, C_DIM, C_DIM, stream);
        launch_gemm(mod_bf, wfT, vfF, nullptr, ROWS, C_DIM, C_DIM, C_DIM, C_DIM, C_DIM, stream);
        launch_gemm(mod_bf, wgT, vgF, nullptr, ROWS, C_DIM, C_DIM, C_DIM, C_DIM, C_DIM, stream);
        hgrn_fi_kernel<<<gridE, 256, 0, stream>>>(vfF, viF);
        hgrn_scan_kernel<<<(BB * C_DIM + 255) / 256, 256, 0, stream>>>(vfF, viF, deltaF);
        hgrn_out_kernel<<<(ROWS * HEADS) / 8, 256, 0, stream>>>(deltaF, vgF, gn, attn_bf);
        launch_gemm(attn_bf, woT, projOut, nullptr, ROWS, C_DIM, C_DIM,
                    C_DIM, C_DIM, C_DIM, stream);
        residual_kernel<<<gridE, 256, 0, stream>>>(Xout, projOut, mods, 2 * C_DIM);

        // ---- gated MLP ----
        ln_mod_kernel<<<ROWS, 256, 0, stream>>>(Xout, gw, gb, mods, 3 * C_DIM, 4 * C_DIM,
                                                nullptr, mod_bf);
        launch_gemm(mod_bf, gateT, uF, nullptr, ROWS, 2 * INTER_D, C_DIM,
                    C_DIM, C_DIM, 2 * INTER_D, stream);
        mlp_fuse_kernel<<<(unsigned)(((size_t)ROWS * INTER_D + 255) / 256), 256, 0, stream>>>(
            uF, fused_bf);
        launch_gemm(fused_bf, downT, deltaF, nullptr, ROWS, C_DIM, INTER_D,
                    INTER_D, INTER_D, C_DIM, stream);
        residual_kernel<<<gridE, 256, 0, stream>>>(Xout, deltaF, mods, 5 * C_DIM);
    }
}
